// AttLocRec_47631187313262
// MI455X (gfx1250) — matrix operand order repaired and checked by reading, not compiled, not run on hardware
//
#include <hip/hip_runtime.h>
#include <math.h>

#define B_    32
#define T_    1600
#define E_    1024
#define D_    1024
#define A_    512
#define C_    10
#define FILT_ 100
#define KW_   (2 * FILT_ + 1)   // 201
#define TCH_  4                 // t-chunks for context partials
#define TSZ_  (T_ / TCH_)       // 400

#define MT_   64                // M rows per workgroup in score GEMM
#define KC_   256               // K chunk staged in LDS per stage
#define KPAD_ 8                 // bf16 pad per LDS row (16B -> shifts bank group/row)

typedef __attribute__((ext_vector_type(16))) __bf16 v16bf;
typedef __attribute__((ext_vector_type(8)))  __bf16 v8bf;
typedef __attribute__((ext_vector_type(8)))  float  v8f;

__device__ __forceinline__ unsigned short f2bf(float f) {
    unsigned int u = __float_as_uint(f);
    unsigned int r = u + 0x7FFFu + ((u >> 16) & 1u);   // round-to-nearest-even
    return (unsigned short)(r >> 16);
}

__device__ __forceinline__ float sigmoidf(float x) {
    return 1.0f / (1.0f + __expf(-x));
}

// ---------------------------------------------------------------------------
// K0: W_enc (A_, E_) fp32 -> bf16 (row-major, becomes WMMA B columns directly)
// ---------------------------------------------------------------------------
__global__ __launch_bounds__(256) void cvt_wenc_kernel(const float* __restrict__ W,
                                                       unsigned short* __restrict__ Wb) {
    int i = (blockIdx.x * 256 + threadIdx.x) * 4;
    float4 v = *(const float4*)&W[i];
    uint2 p;
    p.x = (unsigned int)f2bf(v.x) | ((unsigned int)f2bf(v.y) << 16);
    p.y = (unsigned int)f2bf(v.z) | ((unsigned int)f2bf(v.w) << 16);
    *(uint2*)&Wb[i] = p;
}

// ---------------------------------------------------------------------------
// K1: location conv (B,1,T)->(B,C,T), relu, global max over T -> att_feat(B,C)
// ---------------------------------------------------------------------------
__global__ __launch_bounds__(256) void conv_kernel(const float* __restrict__ att_prev,
                                                   const float* __restrict__ conv_w,
                                                   float* __restrict__ att_feat) {
    __shared__ float shin[T_ + 2 * FILT_];
    __shared__ float shw[KW_];
    __shared__ float red[256];
    const int b = blockIdx.x / C_;
    const int c = blockIdx.x % C_;
    const int tid = threadIdx.x;

    for (int i = tid; i < T_ + 2 * FILT_; i += 256) {
        int t = i - FILT_;
        shin[i] = (t >= 0 && t < T_) ? att_prev[(size_t)b * T_ + t] : 0.0f;
    }
    for (int i = tid; i < KW_; i += 256) shw[i] = conv_w[(size_t)c * KW_ + i];
    __syncthreads();

    float m = 0.0f;  // relu floor: max(relu(x)) == max(0, max(x))
    for (int t = tid; t < T_; t += 256) {
        float acc = 0.0f;
        #pragma unroll 4
        for (int k = 0; k < KW_; ++k) acc += shin[t + k] * shw[k];
        m = fmaxf(m, acc);
    }
    red[tid] = m;
    __syncthreads();
    for (int s = 128; s > 0; s >>= 1) {
        if (tid < s) red[tid] = fmaxf(red[tid], red[tid + s]);
        __syncthreads();
    }
    if (tid == 0) att_feat[(size_t)b * C_ + c] = red[0];
}

// ---------------------------------------------------------------------------
// K2: LSTM cell + dec projection; emits h_new, c_new and fused score bias
//     svec[b,a] = h_new[b,a] + dec_proj[b,a] + b_enc[a]
// ---------------------------------------------------------------------------
__global__ __launch_bounds__(256) void lstm_kernel(const float* __restrict__ att_feat,
                                                   const float* __restrict__ att_h,
                                                   const float* __restrict__ att_c,
                                                   const float* __restrict__ dec_z,
                                                   const float* __restrict__ W_ih,
                                                   const float* __restrict__ W_hh,
                                                   const float* __restrict__ W_dec,
                                                   const float* __restrict__ b_enc,
                                                   float* __restrict__ svec,
                                                   float* __restrict__ h_out,
                                                   float* __restrict__ c_out) {
    __shared__ float shh[A_];
    __shared__ float shz[D_];
    __shared__ float shg[4 * A_];
    __shared__ float shf[C_];
    const int b = blockIdx.x;
    const int tid = threadIdx.x;

    for (int i = tid; i < A_; i += 256) shh[i] = att_h[(size_t)b * A_ + i];
    for (int i = tid; i < D_; i += 256) shz[i] = dec_z[(size_t)b * D_ + i];
    if (tid < C_) shf[tid] = att_feat[(size_t)b * C_ + tid];
    __syncthreads();

    for (int j = tid; j < 4 * A_; j += 256) {
        float g = 0.0f;
        const float* wi = &W_ih[(size_t)j * C_];
        #pragma unroll
        for (int c = 0; c < C_; ++c) g += shf[c] * wi[c];
        const float* wh = &W_hh[(size_t)j * A_];
        #pragma unroll 4
        for (int a = 0; a < A_; ++a) g += shh[a] * wh[a];
        shg[j] = g;
    }
    __syncthreads();

    for (int a = tid; a < A_; a += 256) {
        float ig = sigmoidf(shg[a]);
        float fg = sigmoidf(shg[A_ + a]);
        float gg = tanhf(shg[2 * A_ + a]);
        float og = sigmoidf(shg[3 * A_ + a]);
        float cn = fg * att_c[(size_t)b * A_ + a] + ig * gg;
        float hn = og * tanhf(cn);
        c_out[(size_t)b * A_ + a] = cn;
        h_out[(size_t)b * A_ + a] = hn;
        float dp = 0.0f;
        const float* wd = &W_dec[(size_t)a * D_];
        #pragma unroll 4
        for (int e = 0; e < D_; ++e) dp += shz[e] * wd[e];
        svec[(size_t)b * A_ + a] = hn + dp + b_enc[a];
    }
}

// ---------------------------------------------------------------------------
// K3: fused score GEMM. Block = 64 rows of flattened (b,t) x all N=512.
// Each of 8 waves computes a 64x64 output block: 4 M-subtiles x 4 N-subtiles
// (16 accumulator tiles) -> 16 WMMAs per 8 ds + 8 global B128 loads per K step.
// A staged fp32->bf16 in LDS in K-chunks of 256 (33.8 KB, conflict-free rows).
// Epilogue: e[b,t] = sum_a tanh(acc + svec[b,a]) * W_g[a] + b_g, masked.
// ---------------------------------------------------------------------------
__global__ __launch_bounds__(256) void score_gemm_kernel(const float* __restrict__ enc,
                                                         const unsigned short* __restrict__ Wb,
                                                         const float* __restrict__ svec,
                                                         const float* __restrict__ Wg,
                                                         const float* __restrict__ bg,
                                                         const int* __restrict__ lens,
                                                         float* __restrict__ e_out) {
    __shared__ __align__(16) unsigned short As[MT_][KC_ + KPAD_];  // 64 x 264 bf16
    __shared__ float eacc[MT_];
    const int tid  = threadIdx.x;
    const int row0 = blockIdx.x * MT_;       // flattened (b*T + t); T%64==0 -> single b
    const int b    = row0 / T_;
    const int t0   = row0 % T_;

    const int wave  = tid >> 5;
    const int lane  = tid & 31;
    const int ln    = lane & 15;   // A row-in-subtile / B,D column
    const int lhi   = lane >> 4;   // half-wave: K sub-range / D row half
    const int nbase = wave * 64;

    if (tid < MT_) eacc[tid] = 0.0f;

    v8f zero = {};
    v8f acc[4][4];
    #pragma unroll
    for (int mt = 0; mt < 4; ++mt)
        #pragma unroll
        for (int nt = 0; nt < 4; ++nt) acc[mt][nt] = zero;

    for (int k0 = 0; k0 < E_; k0 += KC_) {
        __syncthreads();   // previous stage fully consumed (also covers eacc init)
        // stage 64 x 256 fp32 -> bf16 (float4 loads, uint2 packed stores)
        for (int p = tid; p < MT_ * (KC_ / 4); p += 256) {
            int row = p >> 6;             // KC_/4 == 64 quads per row
            int col = (p & 63) * 4;
            float4 v = *(const float4*)&enc[(size_t)(row0 + row) * E_ + k0 + col];
            uint2 u;
            u.x = (unsigned int)f2bf(v.x) | ((unsigned int)f2bf(v.y) << 16);
            u.y = (unsigned int)f2bf(v.z) | ((unsigned int)f2bf(v.w) << 16);
            *(uint2*)&As[row][col] = u;
        }
        __syncthreads();

        for (int kk = 0; kk < KC_; kk += 32) {
            // A fragments: lane holds row mt*16+ln, K = kk+{lhi*8..+7} and +16
            v16bf af[4];
            #pragma unroll
            for (int mt = 0; mt < 4; ++mt) {
                const v8bf alo = *(const v8bf*)&As[mt * 16 + ln][kk + lhi * 8];
                const v8bf ahi = *(const v8bf*)&As[mt * 16 + ln][kk + lhi * 8 + 16];
                #pragma unroll
                for (int i = 0; i < 8; ++i) { af[mt][i] = alo[i]; af[mt][i + 8] = ahi[i]; }
            }
            // B fragments: lane holds column ncol, K = kk+{lhi*8..+7} and +16
            const int kb = k0 + kk + lhi * 8;
            v16bf bf[4];
            #pragma unroll
            for (int nt = 0; nt < 4; ++nt) {
                const int ncol = nbase + nt * 16 + ln;
                const v8bf* bp = (const v8bf*)&Wb[(size_t)ncol * E_ + kb];
                const v8bf b0 = bp[0], b1 = bp[2];
                #pragma unroll
                for (int i = 0; i < 8; ++i) { bf[nt][i] = b0[i]; bf[nt][i + 8] = b1[i]; }
            }
            #pragma unroll
            for (int mt = 0; mt < 4; ++mt)
                #pragma unroll
                for (int nt = 0; nt < 4; ++nt)
                    acc[mt][nt] = __builtin_amdgcn_wmma_f32_16x16x32_bf16(
                        false, af[mt], false, bf[nt], (short)0, acc[mt][nt], false, false);
        }
    }

    // epilogue: tanh + dot with W_g. D layout: lane ln = column, VGPR j = row j+8*lhi.
    float bias[4], wg[4];
    #pragma unroll
    for (int nt = 0; nt < 4; ++nt) {
        int a = nbase + nt * 16 + ln;
        bias[nt] = svec[(size_t)b * A_ + a];
        wg[nt] = Wg[a];
    }
    #pragma unroll
    for (int mt = 0; mt < 4; ++mt) {
        float pr[8];
        #pragma unroll
        for (int j = 0; j < 8; ++j) pr[j] = 0.0f;
        #pragma unroll
        for (int nt = 0; nt < 4; ++nt)
            #pragma unroll
            for (int j = 0; j < 8; ++j)
                pr[j] += tanhf(acc[mt][nt][j] + bias[nt]) * wg[nt];
        // reduce across the 16 columns of this wave's half (xor stays in-half)
        #pragma unroll
        for (int j = 0; j < 8; ++j) {
            float v = pr[j];
            v += __shfl_xor(v, 1, 32);
            v += __shfl_xor(v, 2, 32);
            v += __shfl_xor(v, 4, 32);
            v += __shfl_xor(v, 8, 32);
            if (ln == 0) atomicAdd(&eacc[mt * 16 + lhi * 8 + j], v);  // ds_add_f32
        }
    }
    __syncthreads();

    if (tid < MT_) {
        int t = t0 + tid;
        float e = eacc[tid] + bg[0];
        if (t >= lens[b]) e = -3.402823466e38f;   // finfo(f32).min
        e_out[(size_t)row0 + tid] = e;
    }
}

// ---------------------------------------------------------------------------
// K4: masked softmax(2*e) over T, in place in the w output region
// ---------------------------------------------------------------------------
__global__ __launch_bounds__(256) void softmax_kernel(float* __restrict__ w) {
    __shared__ float red[256];
    const int b = blockIdx.x;
    const int tid = threadIdx.x;
    float* row = w + (size_t)b * T_;

    float m = -3.402823466e38f;
    for (int t = tid; t < T_; t += 256) m = fmaxf(m, row[t]);
    red[tid] = m;
    __syncthreads();
    for (int s = 128; s > 0; s >>= 1) {
        if (tid < s) red[tid] = fmaxf(red[tid], red[tid + s]);
        __syncthreads();
    }
    m = red[0];
    __syncthreads();

    float sum = 0.0f;
    for (int t = tid; t < T_; t += 256) sum += __expf(2.0f * (row[t] - m));
    red[tid] = sum;
    __syncthreads();
    for (int s = 128; s > 0; s >>= 1) {
        if (tid < s) red[tid] += red[tid + s];
        __syncthreads();
    }
    float inv = 1.0f / red[0];
    for (int t = tid; t < T_; t += 256) row[t] = __expf(2.0f * (row[t] - m)) * inv;
}

// ---------------------------------------------------------------------------
// K5a: ctx partials over T chunks (deterministic, no fp atomics)
// ---------------------------------------------------------------------------
__global__ __launch_bounds__(256) void ctx_partial_kernel(const float* __restrict__ enc,
                                                          const float* __restrict__ w,
                                                          float* __restrict__ part) {
    __shared__ float shw[TSZ_];
    const int b  = blockIdx.z;
    const int ch = blockIdx.y;
    const int e  = blockIdx.x * 256 + threadIdx.x;
    const int t0 = ch * TSZ_;

    for (int i = threadIdx.x; i < TSZ_; i += 256) shw[i] = w[(size_t)b * T_ + t0 + i];
    __syncthreads();

    float sum = 0.0f;
    const float* ep = enc + ((size_t)b * T_ + t0) * E_ + e;
    #pragma unroll 4
    for (int tt = 0; tt < TSZ_; ++tt) sum += shw[tt] * ep[(size_t)tt * E_];
    part[((size_t)ch * B_ + b) * E_ + e] = sum;
}

// K5b: reduce partials -> ctx
__global__ __launch_bounds__(256) void ctx_reduce_kernel(const float* __restrict__ part,
                                                         float* __restrict__ ctx) {
    int i = blockIdx.x * 256 + threadIdx.x;
    float s = 0.0f;
    #pragma unroll
    for (int ch = 0; ch < TCH_; ++ch) s += part[(size_t)ch * B_ * E_ + i];
    ctx[i] = s;
}

// ---------------------------------------------------------------------------
extern "C" void kernel_launch(void* const* d_in, const int* in_sizes, int n_in,
                              void* d_out, int out_size, void* d_ws, size_t ws_size,
                              hipStream_t stream) {
    (void)in_sizes; (void)n_in; (void)out_size; (void)ws_size;
    const float* enc      = (const float*)d_in[0];
    const int*   lens     = (const int*)d_in[1];
    const float* dec_z    = (const float*)d_in[2];
    const float* att_prev = (const float*)d_in[3];
    const float* att_h    = (const float*)d_in[4];
    const float* att_c    = (const float*)d_in[5];
    const float* W_enc    = (const float*)d_in[6];
    const float* b_enc    = (const float*)d_in[7];
    const float* W_dec    = (const float*)d_in[8];
    const float* conv_w   = (const float*)d_in[9];
    const float* W_ih     = (const float*)d_in[10];
    const float* W_hh     = (const float*)d_in[11];
    const float* W_g      = (const float*)d_in[12];
    const float* b_g      = (const float*)d_in[13];

    float* out  = (float*)d_out;
    float* ctx  = out;                         // (B, E)
    float* wout = ctx + B_ * E_;               // (B, T)  e, then softmaxed in place
    float* hout = wout + B_ * T_;              // (B, A)
    float* cout = hout + B_ * A_;              // (B, A)

    // workspace layout
    unsigned short* Wb = (unsigned short*)d_ws;                 // A_*E_ bf16 (1 MB)
    float* svec     = (float*)((char*)d_ws + (size_t)A_ * E_ * sizeof(unsigned short));
    float* att_feat = svec + B_ * A_;                           // B_*C_
    float* ctx_part = att_feat + ((B_ * C_ + 63) & ~63);        // TCH_*B_*E_

    cvt_wenc_kernel<<<(A_ * E_) / 1024, 256, 0, stream>>>(W_enc, Wb);
    conv_kernel<<<B_ * C_, 256, 0, stream>>>(att_prev, conv_w, att_feat);
    lstm_kernel<<<B_, 256, 0, stream>>>(att_feat, att_h, att_c, dec_z,
                                        W_ih, W_hh, W_dec, b_enc, svec, hout, cout);
    score_gemm_kernel<<<(B_ * T_) / MT_, 256, 0, stream>>>(enc, Wb, svec, W_g, b_g,
                                                           lens, wout);
    softmax_kernel<<<B_, 256, 0, stream>>>(wout);
    ctx_partial_kernel<<<dim3(E_ / 256, TCH_, B_), 256, 0, stream>>>(enc, wout, ctx_part);
    ctx_reduce_kernel<<<(B_ * E_) / 256, 256, 0, stream>>>(ctx_part, ctx);
}
